// CausalSelfAttention_88089779241159
// MI455X (gfx1250) — matrix operand order repaired and checked by reading, not compiled, not run on hardware
//
#include <hip/hip_runtime.h>

typedef __attribute__((ext_vector_type(16))) __bf16       v16bf;
typedef __attribute__((ext_vector_type(8)))  float        v8f;
typedef __attribute__((ext_vector_type(4)))  unsigned int u32x4;

#define C_EMBD 1024
#define T_SEQ  2048
#define N_B    4
#define N_H    16
#define D_H    64
#define BT     (N_B * T_SEQ)

// fold 1/sqrt(64) * log2(e) into Q so softmax uses exp2 (v_exp_f32)
#define Q_SCALE 0.18033688011112042f

union BF16x16 {
  u32x4  q[2];
  __bf16 s[16];
  v16bf  v;
};

// load two 16-byte chunks (K-split table-order pattern, used for both A and B operands)
__device__ __forceinline__ v16bf ld2x16(const __bf16* p0, const __bf16* p1) {
  BF16x16 t;
  t.q[0] = *(const u32x4*)p0;
  t.q[1] = *(const u32x4*)p1;
  return t.v;
}

#define WMMA_BF16(a, b, c) \
  __builtin_amdgcn_wmma_f32_16x16x32_bf16(false, (a), false, (b), (short)0, (c), false, false)

// ---------------------------------------------------------------- converts
__global__ void cvt_x_kernel(const float* __restrict__ x, __bf16* __restrict__ xb, int n) {
  int i = blockIdx.x * 256 + threadIdx.x;
  if (i < n) xb[i] = (__bf16)x[i];
}

// W is (out n, in k) row-major fp32; produce Wt[n][k] bf16 (n-major) so the
// WMMA B-operand (lane = n, per-lane 16 k in table order) is two 16B loads.
__global__ void cvt_wt_kernel(const float* __restrict__ w, __bf16* __restrict__ wt) {
  int i = blockIdx.x * 256 + threadIdx.x;        // i = n*1024 + k
  int n = i >> 10, k = i & 1023;
  wt[n * C_EMBD + k] = (__bf16)w[i];
}

// ---------------------------------------------------------------- QKV GEMM
// y[m,n] = sum_k xb[m,k] * Wt[k,n]; stored as [B,H,T,Dh] bf16 (m=b*T+t, n=h*64+d)
// block = 256 thr (8 waves); wave tile 32(m) x 64(n); block tile 64 x 256
__global__ void gemm_qkv_kernel(const __bf16* __restrict__ xb,
                                const __bf16* __restrict__ wt,
                                __bf16* __restrict__ out, float scale) {
  const int wave = threadIdx.x >> 5, lane = threadIdx.x & 31;
  const int l16 = lane & 15, hl = lane >> 4;
  const int m0 = blockIdx.x * 64 + (wave & 1) * 32;
  const int n0 = blockIdx.y * 256 + (wave >> 1) * 64;

  v8f acc[2][4] = {};
  for (int kb = 0; kb < C_EMBD; kb += 32) {
    v16bf a[2], b[4];
#pragma unroll
    for (int im = 0; im < 2; ++im) {
      const __bf16* pa = xb + (m0 + im * 16 + l16) * C_EMBD + kb + hl * 8;
      a[im] = ld2x16(pa, pa + 16);
    }
#pragma unroll
    for (int in = 0; in < 4; ++in) {
      const __bf16* pb = wt + (n0 + in * 16 + l16) * C_EMBD + kb + hl * 8;
      b[in] = ld2x16(pb, pb + 16);
    }
    // warm L0 one K-step ahead (global_prefetch_b8)
    __builtin_prefetch(xb + (m0 + l16) * C_EMBD + kb + 32 + hl * 8, 0, 1);
    __builtin_prefetch(wt + (kb + 32 + lane) * C_EMBD + n0, 0, 1);
#pragma unroll
    for (int im = 0; im < 2; ++im)
#pragma unroll
      for (int in = 0; in < 4; ++in)
        acc[im][in] = WMMA_BF16(a[im], b[in], acc[im][in]);
  }
#pragma unroll
  for (int im = 0; im < 2; ++im)
#pragma unroll
    for (int in = 0; in < 4; ++in)
#pragma unroll
      for (int r = 0; r < 8; ++r) {
        int m = m0 + im * 16 + r + hl * 8;
        int n = n0 + in * 16 + l16;
        int bb = m >> 11, t = m & 2047;   // T = 2048
        int h = n >> 6, d = n & 63;
        out[(((bb * N_H + h) * T_SEQ) + t) * D_H + d] = (__bf16)(acc[im][in][r] * scale);
      }
}

// ------------------------------------------------------------ flash attention
// grid.x = (B*H) * (T/64); block = 128 thr (4 waves); 1 wave = 16 queries.
// S^T = K * Q^T  (lane = query), online softmax per lane, O = P * V with the
// S^T C-layout feeding the P A-operand directly (no shuffles).
__global__ void attn_kernel(const __bf16* __restrict__ Qm,
                            const __bf16* __restrict__ Km,
                            const __bf16* __restrict__ Vm,
                            __bf16* __restrict__ out) {
  const int bh = blockIdx.x >> 5;          // T/64 = 32 q-blocks
  const int qblk = blockIdx.x & 31;
  const int b = bh >> 4, h = bh & 15;
  const int base = bh * (T_SEQ * D_H);
  const int wave = threadIdx.x >> 5, lane = threadIdx.x & 31;
  const int l16 = lane & 15, hl = lane >> 4;
  const int q_base = qblk * 64 + wave * 16;
  const int qi = q_base + l16;             // this lane's query (both halves)

  __shared__ float af[4][16];

  // Q^T B-operand: lane = query (q_base + l16), per-lane 16 dh in table order.
  // Loaded once, reused across the whole key loop.
  v16bf bq[2];
#pragma unroll
  for (int c = 0; c < 2; ++c) {
    const __bf16* pq = Qm + base + (q_base + l16) * D_H + c * 32 + hl * 8;
    bq[c] = ld2x16(pq, pq + 16);
  }

  v8f accO[4] = {};
  float mrun = -3.0e38f, lrun = 0.0f;

  for (int s0 = 0; s0 < q_base + 16; s0 += 32) {
    // S^T tiles: tt=0 -> keys s0..s0+15, tt=1 -> keys s0+16..s0+31
    v8f st[2];
#pragma unroll
    for (int tt = 0; tt < 2; ++tt) {
      const __bf16* pk = Km + base + (s0 + tt * 16 + l16) * D_H;
      v16bf a0 = ld2x16(pk + hl * 8, pk + 16 + hl * 8);        // dh 0..31
      v16bf a1 = ld2x16(pk + 32 + hl * 8, pk + 48 + hl * 8);   // dh 32..63
      v8f c = {};
      c = WMMA_BF16(a0, bq[0], c);
      c = WMMA_BF16(a1, bq[1], c);
      st[tt] = c;
    }
    // causal mask + tile max (lane holds 16 keys of this 32-step for query qi)
    float s[16];
#pragma unroll
    for (int tt = 0; tt < 2; ++tt)
#pragma unroll
      for (int r = 0; r < 8; ++r) {
        int key = s0 + tt * 16 + r + hl * 8;
        s[tt * 8 + r] = (key <= qi) ? st[tt][r] : -3.0e38f;
      }
    float tm = s[0];
#pragma unroll
    for (int i = 1; i < 16; ++i) tm = fmaxf(tm, s[i]);
    tm = fmaxf(tm, __shfl_xor(tm, 16));

    float mnew = fmaxf(mrun, tm);
    float alpha = __builtin_amdgcn_exp2f(mrun - mnew);
    float psum = 0.0f;
    BF16x16 up;
#pragma unroll
    for (int i = 0; i < 16; ++i) {
      float p = __builtin_amdgcn_exp2f(s[i] - mnew);
      psum += p;
      up.s[i] = (__bf16)p;                 // v_cvt to bf16, no bit-twiddling
    }
    psum += __shfl_xor(psum, 16);
    lrun = lrun * alpha + psum;
    mrun = mnew;

    // broadcast per-query alpha to the O-tile row dimension via per-wave LDS
    if (lane < 16) af[wave][l16] = alpha;
    __builtin_amdgcn_wave_barrier();
    float ar[8];
#pragma unroll
    for (int r = 0; r < 8; ++r) ar[r] = af[wave][hl * 8 + r];
    __builtin_amdgcn_wave_barrier();

    v16bf pA = up.v;   // S^T C-layout == P A-operand layout: no shuffles

    // O = P * V : B-operand lane = d (j*16 + l16), per-lane 16 keys in table order
#pragma unroll
    for (int j = 0; j < 4; ++j) {
      BF16x16 uv;
#pragma unroll
      for (int e = 0; e < 16; ++e)
        uv.s[e] = Vm[base + (s0 + (e < 8 ? hl * 8 + e : 16 + hl * 8 + e - 8)) * D_H + j * 16 + l16];
      v16bf bv = uv.v;
      v8f c;
#pragma unroll
      for (int r = 0; r < 8; ++r) c[r] = accO[j][r] * ar[r];
      c = WMMA_BF16(pA, bv, c);
      accO[j] = c;
    }
  }

  // normalize by row sum (broadcast 1/l like alpha) and store bf16 [B,T,C]
  float inv = __builtin_amdgcn_rcpf(lrun);
  if (lane < 16) af[wave][l16] = inv;
  __builtin_amdgcn_wave_barrier();
  float ir[8];
#pragma unroll
  for (int r = 0; r < 8; ++r) ir[r] = af[wave][hl * 8 + r];
#pragma unroll
  for (int j = 0; j < 4; ++j)
#pragma unroll
    for (int r = 0; r < 8; ++r) {
      int q = q_base + r + hl * 8;
      int d = j * 16 + l16;
      out[(b * T_SEQ + q) * C_EMBD + h * D_H + d] = (__bf16)(accO[j][r] * ir[r]);
    }
}

// ---------------------------------------------------------- output projection
__global__ void gemm_oproj_kernel(const __bf16* __restrict__ ab,
                                  const __bf16* __restrict__ wt,
                                  float* __restrict__ out) {
  const int wave = threadIdx.x >> 5, lane = threadIdx.x & 31;
  const int l16 = lane & 15, hl = lane >> 4;
  const int m0 = blockIdx.x * 64 + (wave & 1) * 32;
  const int n0 = blockIdx.y * 256 + (wave >> 1) * 64;

  v8f acc[2][4] = {};
  for (int kb = 0; kb < C_EMBD; kb += 32) {
    v16bf a[2], b[4];
#pragma unroll
    for (int im = 0; im < 2; ++im) {
      const __bf16* pa = ab + (m0 + im * 16 + l16) * C_EMBD + kb + hl * 8;
      a[im] = ld2x16(pa, pa + 16);
    }
#pragma unroll
    for (int in = 0; in < 4; ++in) {
      const __bf16* pb = wt + (n0 + in * 16 + l16) * C_EMBD + kb + hl * 8;
      b[in] = ld2x16(pb, pb + 16);
    }
    __builtin_prefetch(ab + (m0 + l16) * C_EMBD + kb + 32 + hl * 8, 0, 1);
    __builtin_prefetch(wt + (kb + 32 + lane) * C_EMBD + n0, 0, 1);
#pragma unroll
    for (int im = 0; im < 2; ++im)
#pragma unroll
      for (int in = 0; in < 4; ++in)
        acc[im][in] = WMMA_BF16(a[im], b[in], acc[im][in]);
  }
#pragma unroll
  for (int im = 0; im < 2; ++im)
#pragma unroll
    for (int in = 0; in < 4; ++in)
#pragma unroll
      for (int r = 0; r < 8; ++r) {
        int m = m0 + im * 16 + r + hl * 8;
        int n = n0 + in * 16 + l16;
        out[m * C_EMBD + n] = acc[im][in][r];
      }
}

// ------------------------------------------------------------------- launch
extern "C" void kernel_launch(void* const* d_in, const int* in_sizes, int n_in,
                              void* d_out, int out_size, void* d_ws, size_t ws_size,
                              hipStream_t stream) {
  const float* x  = (const float*)d_in[0];
  const float* wq = (const float*)d_in[1];
  const float* wk = (const float*)d_in[2];
  const float* wv = (const float*)d_in[3];
  const float* wo = (const float*)d_in[4];
  float* outp = (float*)d_out;

  char* ws = (char*)d_ws;
  __bf16* xb  = (__bf16*)ws;                          // 16 MB (reused as attn-out)
  __bf16* wtq = (__bf16*)(ws + (size_t)16777216);
  __bf16* wtk = wtq + 1048576;                        // 2 MB each
  __bf16* wtv = wtk + 1048576;
  __bf16* wto = wtv + 1048576;
  __bf16* Qb  = (__bf16*)(ws + (size_t)25165824);     // 16 MB each
  __bf16* Kb  = Qb + (size_t)BT * D_H * N_H;          // 8388608 elems
  __bf16* Vb  = Kb + (size_t)BT * D_H * N_H;

  // 1) fp32 -> bf16 (x) and fp32 -> bf16 transposed (weights)
  cvt_x_kernel<<<(BT * C_EMBD) / 256, 256, 0, stream>>>(x, xb, BT * C_EMBD);
  cvt_wt_kernel<<<(C_EMBD * C_EMBD) / 256, 256, 0, stream>>>(wq, wtq);
  cvt_wt_kernel<<<(C_EMBD * C_EMBD) / 256, 256, 0, stream>>>(wk, wtk);
  cvt_wt_kernel<<<(C_EMBD * C_EMBD) / 256, 256, 0, stream>>>(wv, wtv);
  cvt_wt_kernel<<<(C_EMBD * C_EMBD) / 256, 256, 0, stream>>>(wo, wto);

  // 2) QKV projections (Q carries softmax scale * log2e)
  dim3 ggrid(BT / 64, C_EMBD / 256);
  gemm_qkv_kernel<<<ggrid, 256, 0, stream>>>(xb, wtq, Qb, Q_SCALE);
  gemm_qkv_kernel<<<ggrid, 256, 0, stream>>>(xb, wtk, Kb, 1.0f);
  gemm_qkv_kernel<<<ggrid, 256, 0, stream>>>(xb, wtv, Vb, 1.0f);

  // 3) flash attention -> bf16 [B,T,C] (reuses xb buffer)
  attn_kernel<<<(N_B * N_H) * (T_SEQ / 64), 128, 0, stream>>>(Qb, Kb, Vb, xb);

  // 4) output projection -> fp32
  gemm_oproj_kernel<<<ggrid, 256, 0, stream>>>(xb, wto, outp);
}
